// TangentLin_6390911336489
// MI455X (gfx1250) — matrix operand order repaired and checked by reading, not compiled, not run on hardware
//
#include <hip/hip_runtime.h>

typedef __attribute__((ext_vector_type(16))) __bf16 v16bf;
typedef __attribute__((ext_vector_type(8)))  float  v8f;

#define N_ROWS  100000
#define K_DIM   512
#define OUT_DIM 512
#define BM      128
#define BN      128
#define BK      32
#define KSTEPS  (K_DIM / BK)

union Frag {
    v16bf        v;
    uint4        q[2];
    unsigned int u[8];
    __bf16       h[16];
};

// 256 threads: thread t owns row t/2, 16 consecutive k at (t%2)*16.
__device__ __forceinline__ void load_tile(const float* __restrict__ src,
                                          int rowBase, int rowClamp, int k0,
                                          int tid, float4 t[4]) {
    int r    = tid >> 1;
    int koff = (tid & 1) * 16;
    int row  = rowBase + r;
    if (row > rowClamp) row = rowClamp;       // clamp: valid memory, result discarded
    const float4* p = (const float4*)(src + (size_t)row * K_DIM + k0 + koff);
    t[0] = p[0]; t[1] = p[1]; t[2] = p[2]; t[3] = p[3];
}

__device__ __forceinline__ void cvt_store_tile(const float4 t[4],
                                               __bf16 (*dst)[BK], int tid) {
    int r    = tid >> 1;
    int koff = (tid & 1) * 16;
    union { __bf16 h[16]; uint4 q[2]; } u;
    u.h[ 0] = (__bf16)t[0].x; u.h[ 1] = (__bf16)t[0].y; u.h[ 2] = (__bf16)t[0].z; u.h[ 3] = (__bf16)t[0].w;
    u.h[ 4] = (__bf16)t[1].x; u.h[ 5] = (__bf16)t[1].y; u.h[ 6] = (__bf16)t[1].z; u.h[ 7] = (__bf16)t[1].w;
    u.h[ 8] = (__bf16)t[2].x; u.h[ 9] = (__bf16)t[2].y; u.h[10] = (__bf16)t[2].z; u.h[11] = (__bf16)t[2].w;
    u.h[12] = (__bf16)t[3].x; u.h[13] = (__bf16)t[3].y; u.h[14] = (__bf16)t[3].z; u.h[15] = (__bf16)t[3].w;
    *(uint4*)&dst[r][koff]     = u.q[0];
    *(uint4*)&dst[r][koff + 8] = u.q[1];
}

__global__ __launch_bounds__(256) void cplx_linear_wmma(
        const float* __restrict__ xre, const float* __restrict__ xim,
        const float* __restrict__ Wre, const float* __restrict__ Wim,
        float* __restrict__ out) {

    // Double-buffered tiles: 4 arrays x 2 buffers x 8KB = 64 KB LDS
    __shared__ __bf16 sAre[2][BM][BK];
    __shared__ __bf16 sAim[2][BM][BK];
    __shared__ __bf16 sBre[2][BN][BK];
    __shared__ __bf16 sBim[2][BN][BK];

    const int tid   = threadIdx.x;
    const int col0  = blockIdx.x * BN;   // column tiles vary FASTEST in dispatch:
    const int row0  = blockIdx.y * BM;   // the 4 col-tiles of a row-block reuse x via L2
    const int wave  = tid >> 5;
    const int lane  = tid & 31;
    const int waveM = wave & 3;          // 4 row-groups of 32 rows
    const int waveN = wave >> 2;         // 2 col-groups of 64 cols
    const int lr    = lane & 15;
    const int hi    = lane >> 4;         // half-lane K-split select

    v8f accRe[2][4], accIm[2][4];
#pragma unroll
    for (int mt = 0; mt < 2; ++mt)
#pragma unroll
        for (int nt = 0; nt < 4; ++nt) {
            accRe[mt][nt] = (v8f){0.f,0.f,0.f,0.f,0.f,0.f,0.f,0.f};
            accIm[mt][nt] = (v8f){0.f,0.f,0.f,0.f,0.f,0.f,0.f,0.f};
        }

    // ---- Prologue: fill buffer 0
    float4 txr[4], txi[4];
    load_tile(xre, row0, N_ROWS - 1, 0, tid, txr);
    load_tile(xim, row0, N_ROWS - 1, 0, tid, txi);
    {
        float4 tw[4];
        load_tile(Wre, col0, OUT_DIM - 1, 0, tid, tw);
        cvt_store_tile(tw, sBre[0], tid);
        float4 tw2[4];
        load_tile(Wim, col0, OUT_DIM - 1, 0, tid, tw2);
        cvt_store_tile(tw2, sBim[0], tid);
    }
    cvt_store_tile(txr, sAre[0], tid);
    cvt_store_tile(txi, sAim[0], tid);
    __syncthreads();

#pragma unroll 2
    for (int it = 0; it < KSTEPS; ++it) {
        const int  p    = it & 1;
        const int  q    = p ^ 1;
        const int  k1   = (it + 1) * BK;
        const bool more = (it + 1) < KSTEPS;

        // Issue next x tile's HBM loads BEFORE the WMMA block (latency spans 32 WMMAs)
        if (more) {
            load_tile(xre, row0, N_ROWS - 1, k1, tid, txr);
            load_tile(xim, row0, N_ROWS - 1, k1, tid, txi);
        }
        if (it + 2 < KSTEPS) {           // warm L2 two tiles ahead on the x stream
            int pr = row0 + (tid >> 1);
            if (pr > N_ROWS - 1) pr = N_ROWS - 1;
            const size_t off = (size_t)pr * K_DIM + (it + 2) * BK + (tid & 1) * 16;
            __builtin_prefetch(xre + off, 0, 1);
            __builtin_prefetch(xim + off, 0, 1);
        }

        // --- A fragments: bf16 16x32, lane<16 holds K{0..7,16..23}, lane>=16 K{8..15,24..31}
        Frag aRe[2], aIm[2];
#pragma unroll
        for (int mt = 0; mt < 2; ++mt) {
            const int ar = waveM * 32 + mt * 16 + lr;
            aRe[mt].q[0] = *(const uint4*)&sAre[p][ar][hi * 8];
            aRe[mt].q[1] = *(const uint4*)&sAre[p][ar][16 + hi * 8];
            aIm[mt].q[0] = *(const uint4*)&sAim[p][ar][hi * 8];
            aIm[mt].q[1] = *(const uint4*)&sAim[p][ar][16 + hi * 8];
        }

#pragma unroll
        for (int nt = 0; nt < 4; ++nt) {
            // --- B fragments: bf16 32x16, lane<16 holds K{0..7,16..23} of col, lane>=16 K{8..15,24..31}
            const int bc = waveN * 64 + nt * 16 + lr;
            Frag bRe, bIm, bImN;
            bRe.q[0] = *(const uint4*)&sBre[p][bc][hi * 8];
            bRe.q[1] = *(const uint4*)&sBre[p][bc][16 + hi * 8];
            bIm.q[0] = *(const uint4*)&sBim[p][bc][hi * 8];
            bIm.q[1] = *(const uint4*)&sBim[p][bc][16 + hi * 8];
#pragma unroll
            for (int j = 0; j < 8; ++j)  // -b_im via sign flip (bf16 WMMA NEG[1:0] must be 0)
                bImN.u[j] = bIm.u[j] ^ 0x80008000u;
#pragma unroll
            for (int mt = 0; mt < 2; ++mt) {
                accRe[mt][nt] = __builtin_amdgcn_wmma_f32_16x16x32_bf16(
                    false, aRe[mt].v, false, bRe.v,  (short)0, accRe[mt][nt], false, false);
                accRe[mt][nt] = __builtin_amdgcn_wmma_f32_16x16x32_bf16(
                    false, aIm[mt].v, false, bImN.v, (short)0, accRe[mt][nt], false, false);
                accIm[mt][nt] = __builtin_amdgcn_wmma_f32_16x16x32_bf16(
                    false, aRe[mt].v, false, bIm.v,  (short)0, accIm[mt][nt], false, false);
                accIm[mt][nt] = __builtin_amdgcn_wmma_f32_16x16x32_bf16(
                    false, aIm[mt].v, false, bRe.v,  (short)0, accIm[mt][nt], false, false);
            }
        }

        // Fill the other buffer: W tiles (L2-resident) staged here; x temps converted now,
        // so the HBM-load wait lands after the WMMA block.
        if (more) {
            float4 twr[4], twi[4];
            load_tile(Wre, col0, OUT_DIM - 1, k1, tid, twr);
            load_tile(Wim, col0, OUT_DIM - 1, k1, tid, twi);
            cvt_store_tile(twr, sBre[q], tid);
            cvt_store_tile(twi, sBim[q], tid);
            cvt_store_tile(txr, sAre[q], tid);
            cvt_store_tile(txi, sAim[q], tid);
        }
        __syncthreads();   // single barrier per K-step (double buffer)
    }

    // Epilogue: C/D layout — VGPR r: lane<16 -> (M=r, N=lane), lane>=16 -> (M=r+8, N=lane-16)
    float* outRe = out;
    float* outIm = out + (size_t)N_ROWS * OUT_DIM;
#pragma unroll
    for (int mt = 0; mt < 2; ++mt) {
#pragma unroll
        for (int nt = 0; nt < 4; ++nt) {
            const int gc = col0 + waveN * 64 + nt * 16 + lr;
#pragma unroll
            for (int r = 0; r < 8; ++r) {
                const int gr = row0 + waveM * 32 + mt * 16 + hi * 8 + r;
                if (gr < N_ROWS) {
                    outRe[(size_t)gr * OUT_DIM + gc] = accRe[mt][nt][r];
                    outIm[(size_t)gr * OUT_DIM + gc] = accIm[mt][nt][r];
                }
            }
        }
    }
}

extern "C" void kernel_launch(void* const* d_in, const int* in_sizes, int n_in,
                              void* d_out, int out_size, void* d_ws, size_t ws_size,
                              hipStream_t stream) {
    (void)in_sizes; (void)n_in; (void)out_size; (void)d_ws; (void)ws_size;
    const float* xre = (const float*)d_in[0];
    const float* xim = (const float*)d_in[1];
    const float* Wre = (const float*)d_in[2];
    const float* Wim = (const float*)d_in[3];
    float* out = (float*)d_out;

    // x-fastest dispatch = column tiles of the same row-block are adjacent:
    // x row-block (512 KB) is read once from HBM, 3 more times from L2.
    dim3 grid(OUT_DIM / BN, (N_ROWS + BM - 1) / BM);   // (4, 782)
    dim3 block(256);
    cplx_linear_wmma<<<grid, block, 0, stream>>>(xre, xim, Wre, Wim, out);
}
